// SelfAttentionForJet_3805341024473
// MI455X (gfx1250) — matrix operand order repaired and checked by reading, not compiled, not run on hardware
//
#include <hip/hip_runtime.h>
#include <cstdint>

#define NB    8
#define SEQ   1024
#define DIMM  768
#define NHEAD 12
#define HD    64
#define QKVN  2304

typedef __attribute__((ext_vector_type(16))) _Float16 v16h;
typedef __attribute__((ext_vector_type(8)))  float    v8f;

union Frag { v16h v; uint4 u[2]; };

__device__ inline v8f zero8() {
  v8f z = {0.f,0.f,0.f,0.f,0.f,0.f,0.f,0.f};
  return z;
}

// ---------------------------------------------------------------------------
// CDNA5 async global->LDS staging (ASYNCcnt path).  Falls back to a
// synchronous VGPR round-trip copy when the builtin is unavailable.
// Each call stages 16 f16 (32 B) via two b128 transfers.
// ---------------------------------------------------------------------------
#if __has_builtin(__builtin_amdgcn_global_load_async_to_lds_b128)
#define USE_ASYNC_LDS 1
typedef __attribute__((__vector_size__(4 * sizeof(int)))) int aint4;
typedef aint4 __attribute__((address_space(1)))* gv4p;   // global int4*
typedef aint4 __attribute__((address_space(3)))* lv4p;   // LDS int4*
#endif

__device__ inline void stage32B(const _Float16* gsrc, _Float16* ldst) {
#ifdef USE_ASYNC_LDS
  __builtin_amdgcn_global_load_async_to_lds_b128((gv4p)gsrc, (lv4p)ldst, 0, 0);
  __builtin_amdgcn_global_load_async_to_lds_b128((gv4p)(gsrc + 8),
                                                 (lv4p)(ldst + 8), 0, 0);
#else
  const uint4* s = reinterpret_cast<const uint4*>(gsrc);
  uint4* d = reinterpret_cast<uint4*>(ldst);
  d[0] = s[0];
  d[1] = s[1];
#endif
}

__device__ inline void wait_stage() {
#ifdef USE_ASYNC_LDS
#if __has_builtin(__builtin_amdgcn_s_wait_asynccnt)
  __builtin_amdgcn_s_wait_asynccnt(0);
#else
  asm volatile("s_wait_asynccnt 0x0" ::: "memory");
#endif
#endif
}

// ---------------------------------------------------------------------------
// Cross-lane XOR shuffle within 16-lane rows, VALU-only (v_permlane16_b32)
// instead of ds_bpermute: keeps the LDS pipe free for the P-tile staging.
// ---------------------------------------------------------------------------
#if __has_builtin(__builtin_amdgcn_permlane16)
__device__ inline float xorshuf16(float v, unsigned s0, unsigned s1) {
  int i = __builtin_bit_cast(int, v);
  int r = __builtin_amdgcn_permlane16(i, i, s0, s1, false, false);
  return __builtin_bit_cast(float, r);
}
__device__ inline float red16_max(float v) {
  v = fmaxf(v, xorshuf16(v, 0x67452301u, 0xEFCDAB89u));  // xor 1
  v = fmaxf(v, xorshuf16(v, 0x54761032u, 0xDCFE98BAu));  // xor 2
  v = fmaxf(v, xorshuf16(v, 0x32107654u, 0xBA98FEDCu));  // xor 4
  v = fmaxf(v, xorshuf16(v, 0xFEDCBA98u, 0x76543210u));  // xor 8
  return v;
}
__device__ inline float red16_sum(float v) {
  v += xorshuf16(v, 0x67452301u, 0xEFCDAB89u);
  v += xorshuf16(v, 0x54761032u, 0xDCFE98BAu);
  v += xorshuf16(v, 0x32107654u, 0xBA98FEDCu);
  v += xorshuf16(v, 0xFEDCBA98u, 0x76543210u);
  return v;
}
#else
__device__ inline float red16_max(float v) {
  for (int off = 1; off < 16; off <<= 1) v = fmaxf(v, __shfl_xor(v, off, 32));
  return v;
}
__device__ inline float red16_sum(float v) {
  for (int off = 1; off < 16; off <<= 1) v += __shfl_xor(v, off, 32);
  return v;
}
#endif

// ---------------------------------------------------------------------------
// LayerNorm (fp32 in) -> f16 out.  One block (256 thr) per token, 768 elems.
// ---------------------------------------------------------------------------
__global__ void __launch_bounds__(256)
ln_kernel(const float* __restrict__ x, const float* __restrict__ g,
          const float* __restrict__ b, _Float16* __restrict__ xn) {
  const int row = blockIdx.x;
  const int t   = threadIdx.x;
  const float* xr = x + (size_t)row * DIMM;
  float v0 = xr[t], v1 = xr[t + 256], v2 = xr[t + 512];
  float s = v0 + v1 + v2;
  float q = v0*v0 + v1*v1 + v2*v2;
  #pragma unroll
  for (int off = 1; off < 32; off <<= 1) {
    s += __shfl_xor(s, off, 32);
    q += __shfl_xor(q, off, 32);
  }
  __shared__ float rs[8], rq[8], mush[2];
  const int wid = t >> 5, lane = t & 31;
  if (lane == 0) { rs[wid] = s; rq[wid] = q; }
  __syncthreads();
  if (t == 0) {
    float S = 0.f, Q = 0.f;
    #pragma unroll
    for (int i = 0; i < 8; i++) { S += rs[i]; Q += rq[i]; }
    float mu  = S * (1.0f / DIMM);
    float var = Q * (1.0f / DIMM) - mu * mu;
    mush[0] = mu;
    mush[1] = rsqrtf(var + 1e-5f);
  }
  __syncthreads();
  const float mu = mush[0], ri = mush[1];
  _Float16* o = xn + (size_t)row * DIMM;
  o[t]       = (_Float16)((v0 - mu) * ri * g[t]       + b[t]);
  o[t + 256] = (_Float16)((v1 - mu) * ri * g[t + 256] + b[t + 256]);
  o[t + 512] = (_Float16)((v2 - mu) * ri * g[t + 512] + b[t + 512]);
}

// ---------------------------------------------------------------------------
// Weight transpose + fp32->f16:  dst[ncol*K + kc] = src[kc*Ncols + ncol]
// ---------------------------------------------------------------------------
__global__ void cvtT_kernel(const float* __restrict__ src, _Float16* __restrict__ dst,
                            int K, int Ncols) {
  const size_t total = (size_t)K * (size_t)Ncols;
  for (size_t i = (size_t)blockIdx.x * blockDim.x + threadIdx.x; i < total;
       i += (size_t)gridDim.x * blockDim.x) {
    const int ncol = (int)(i / (size_t)K);
    const int kc   = (int)(i % (size_t)K);
    dst[i] = (_Float16)src[(size_t)kc * Ncols + ncol];
  }
}

// ---------------------------------------------------------------------------
// WMMA GEMM:  C[M x N] = A[M x 768] * Bt[N x 768]^T.  Block tile 128x128,
// 8 waves (2x4), each wave 64x32.  Double-buffered LDS tiles filled by
// async global->LDS (ASYNCcnt) so the copy for chunk k+1 overlaps the 8
// WMMAs of chunk k.
// EPI=0: qkv epilogue (bias + scatter to q / k row-major, v transposed)
// EPI=1: out epilogue (bias + residual, fp32 store)
// ---------------------------------------------------------------------------
template <int EPI>
__global__ void __launch_bounds__(256)
gemm_kernel(const _Float16* __restrict__ A, const _Float16* __restrict__ Bt,
            const float* __restrict__ bias, const float* __restrict__ skip,
            float* __restrict__ outf,
            _Float16* __restrict__ qh, _Float16* __restrict__ kh,
            _Float16* __restrict__ vt) {
  __shared__ _Float16 As[2][128][40];   // [buf][m-row][kc], 80 B rows (16B-mult)
  __shared__ _Float16 Bs[2][128][40];   // [buf][n-col][kc]
  const int t = threadIdx.x, lane = t & 31, wid = t >> 5;
  const int bm = blockIdx.y, bn = blockIdx.x;
  const int waveM = wid & 1, waveN = wid >> 1;
  const int r0 = lane & 15, hi = lane >> 4, rowadd = hi * 8;

  v8f acc[4][2];
  #pragma unroll
  for (int m = 0; m < 4; m++)
    #pragma unroll
    for (int n = 0; n < 2; n++) acc[m][n] = zero8();

  const int ldRow = t >> 1, ldOff = (t & 1) * 16;
  const _Float16* aSrcBase = A  + (size_t)(bm * 128 + ldRow) * DIMM + ldOff;
  const _Float16* bSrcBase = Bt + (size_t)(bn * 128 + ldRow) * DIMM + ldOff;

  // Prologue: stage K-chunk 0 into buffer 0.
  stage32B(aSrcBase, &As[0][ldRow][ldOff]);
  stage32B(bSrcBase, &Bs[0][ldRow][ldOff]);

  const int NIT = DIMM / 32;
  for (int it = 0; it < NIT; ++it) {
    const int p = it & 1;
    wait_stage();
    __syncthreads();
    if (it + 1 < NIT) {                       // prefetch next chunk into p^1
      const int kb2 = (it + 1) * 32;
      stage32B(aSrcBase + kb2, &As[p ^ 1][ldRow][ldOff]);
      stage32B(bSrcBase + kb2, &Bs[p ^ 1][ldRow][ldOff]);
    }

    Frag a[4], bf[2];
    #pragma unroll
    for (int m = 0; m < 4; m++) {
      const int r = waveM * 64 + m * 16 + r0;
      a[m].u[0] = *reinterpret_cast<const uint4*>(&As[p][r][hi * 8]);
      a[m].u[1] = *reinterpret_cast<const uint4*>(&As[p][r][16 + hi * 8]);
    }
    #pragma unroll
    for (int n = 0; n < 2; n++) {
      const int c = waveN * 32 + n * 16 + r0;
      bf[n].u[0] = *reinterpret_cast<const uint4*>(&Bs[p][c][hi * 8]);
      bf[n].u[1] = *reinterpret_cast<const uint4*>(&Bs[p][c][16 + hi * 8]);
    }
    #pragma unroll
    for (int m = 0; m < 4; m++)
      #pragma unroll
      for (int n = 0; n < 2; n++)
        acc[m][n] = __builtin_amdgcn_wmma_f32_16x16x32_f16(
            false, a[m].v, false, bf[n].v, (short)0, acc[m][n], false, false);
    __syncthreads();
  }

  // Epilogue. C layout: col = lane&15, row = j + 8*(lane>=16).
  #pragma unroll
  for (int m = 0; m < 4; m++) {
    #pragma unroll
    for (int n = 0; n < 2; n++) {
      const int gc = bn * 128 + waveN * 32 + n * 16 + r0;
      const float bv = bias[gc];
      #pragma unroll
      for (int j = 0; j < 8; j++) {
        const int gr = bm * 128 + waveM * 64 + m * 16 + rowadd + j;
        const float val = acc[m][n][j] + bv;
        if (EPI == 0) {
          const int which = gc / DIMM;           // 0=q 1=k 2=v
          const int hh    = (gc % DIMM) / HD;
          const int d     = gc % HD;
          const int n_    = gr >> 10;
          const int s_    = gr & 1023;
          const size_t nh = (size_t)(n_ * NHEAD + hh);
          if (which == 0)      qh[(nh * SEQ + s_) * HD + d] = (_Float16)val;
          else if (which == 1) kh[(nh * SEQ + s_) * HD + d] = (_Float16)val;
          else                 vt[(nh * HD + d) * SEQ + s_] = (_Float16)val;
        } else {
          const size_t idx = (size_t)gr * DIMM + gc;
          outf[idx] = val + skip[idx];
        }
      }
    }
  }
}

// ---------------------------------------------------------------------------
// RoPE, in place on q and k (both [n*h][s][64] row-major).
// Rotate channel pairs (d, d+16), d in [0,16); angle from (ph,pw)=(s/32,s%32).
// ---------------------------------------------------------------------------
__global__ void rope_kernel(_Float16* __restrict__ q, _Float16* __restrict__ k) {
  const int idx = blockIdx.x * blockDim.x + threadIdx.x;
  if (idx >= NB * NHEAD * SEQ * 16) return;
  const int d  = idx & 15;
  const int s  = (idx >> 4) & 1023;
  const int nh = idx >> 14;
  const int ph = s >> 5, pw = s & 31;
  const int fi = d & 7;
  const float f   = powf(10000.0f, -(float)fi * 0.125f);
  const float ang = (float)((d < 8) ? ph : pw) * f;
  const float c = __cosf(ang), sn = __sinf(ang);
  const size_t base = ((size_t)nh * SEQ + s) * HD;
  float x1 = (float)q[base + d], x2 = (float)q[base + d + 16];
  q[base + d]      = (_Float16)(x1 * c - x2 * sn);
  q[base + d + 16] = (_Float16)(x2 * c + x1 * sn);
  x1 = (float)k[base + d]; x2 = (float)k[base + d + 16];
  k[base + d]      = (_Float16)(x1 * c - x2 * sn);
  k[base + d + 16] = (_Float16)(x2 * c + x1 * sn);
}

// ---------------------------------------------------------------------------
// Flash attention with 7x7 neighborhood mask.  One block (4 waves) per
// (n,h,64-query tile); each wave owns 16 queries.  K restricted to the
// window-reachable range [64*qt-99, 64*qt+163).  Per K-chunk of 32 keys:
// 4 WMMA for Q*K^T, online softmax via VALU permlane16 butterflies, P staged
// through per-wave LDS into A-operand layout, 4 WMMA for P*V.
// ---------------------------------------------------------------------------
__global__ void __launch_bounds__(128)
attn_kernel(const _Float16* __restrict__ q, const _Float16* __restrict__ k,
            const _Float16* __restrict__ vt, _Float16* __restrict__ o) {
  __shared__ _Float16 Ps[4][16][40];
  const int lane = threadIdx.x & 31, wid = threadIdx.x >> 5;
  const int qt = blockIdx.x & 15;
  const int nh = blockIdx.x >> 4;
  const int h  = nh % NHEAD;
  const int n_ = nh / NHEAD;
  const int qbase = qt * 64 + wid * 16;
  const int c16 = lane & 15, g = lane >> 4, rowadd = g * 8;

  // Q fragments (A-operand): lane = query row, two 16-B loads per frag.
  Frag aQ[2];
  {
    const _Float16* qp = q + ((size_t)nh * SEQ + qbase + c16) * HD;
    #pragma unroll
    for (int c = 0; c < 2; c++) {
      aQ[c].u[0] = *reinterpret_cast<const uint4*>(qp + c * 32 + g * 8);
      aQ[c].u[1] = *reinterpret_cast<const uint4*>(qp + c * 32 + 16 + g * 8);
    }
  }

  float mj[8], lj[8];
  #pragma unroll
  for (int j = 0; j < 8; j++) { mj[j] = -3.0e38f; lj[j] = 0.f; }
  v8f oacc[4];
  #pragma unroll
  for (int d = 0; d < 4; d++) oacc[d] = zero8();

  int klo = qt * 64 - 99; if (klo < 0) klo = 0; klo &= ~31;
  int kend = qt * 64 + 163; if (kend > SEQ) kend = SEQ;

  for (int kb = klo; kb < kend; kb += 32) {
    // --- scores: S[16x32] = Q * K^T (2 d-chunks x 2 key tiles) ---
    v8f sc[2]; sc[0] = zero8(); sc[1] = zero8();
    #pragma unroll
    for (int dc = 0; dc < 2; dc++) {
      Frag bK[2];
      #pragma unroll
      for (int nt = 0; nt < 2; nt++) {
        const _Float16* kp =
            k + ((size_t)nh * SEQ + kb + nt * 16 + c16) * HD + dc * 32 + g * 8;
        bK[nt].u[0] = *reinterpret_cast<const uint4*>(kp);
        bK[nt].u[1] = *reinterpret_cast<const uint4*>(kp + 16);
      }
      sc[0] = __builtin_amdgcn_wmma_f32_16x16x32_f16(
          false, aQ[dc].v, false, bK[0].v, (short)0, sc[0], false, false);
      sc[1] = __builtin_amdgcn_wmma_f32_16x16x32_f16(
          false, aQ[dc].v, false, bK[1].v, (short)0, sc[1], false, false);
    }

    // --- scale + mask + online softmax (per C-layout row) ---
    float al[8];
    const int key0 = kb + c16, key1 = kb + 16 + c16;
    const int k0ph = key0 >> 5, k0pw = key0 & 31;
    const int k1ph = key1 >> 5, k1pw = key1 & 31;
    #pragma unroll
    for (int j = 0; j < 8; j++) {
      const int qi = qbase + rowadd + j;
      const int qph = qi >> 5, qpw = qi & 31;
      float v0 = sc[0][j] * 0.125f;
      float v1 = sc[1][j] * 0.125f;
      int dh0 = qph - k0ph; if (dh0 < 0) dh0 = -dh0;
      int dw0 = qpw - k0pw; if (dw0 < 0) dw0 = -dw0;
      int dh1 = qph - k1ph; if (dh1 < 0) dh1 = -dh1;
      int dw1 = qpw - k1pw; if (dw1 < 0) dw1 = -dw1;
      if (dh0 > 3 || dw0 > 3) v0 = -1e30f;
      if (dh1 > 3 || dw1 > 3) v1 = -1e30f;
      const float rm = red16_max(fmaxf(v0, v1));
      const float mnew = fmaxf(mj[j], rm);
      const float a  = __expf(mj[j] - mnew);
      const float p0 = __expf(v0 - mnew);
      const float p1 = __expf(v1 - mnew);
      const float rsum = red16_sum(p0 + p1);
      lj[j] = lj[j] * a + rsum;
      mj[j] = mnew;
      al[j] = a;
      Ps[wid][rowadd + j][c16]      = (_Float16)p0;
      Ps[wid][rowadd + j][16 + c16] = (_Float16)p1;
    }
    #pragma unroll
    for (int d = 0; d < 4; d++)
      #pragma unroll
      for (int j = 0; j < 8; j++) oacc[d][j] *= al[j];
    __syncthreads();

    // --- O += P * V (4 d-chunks; V transposed so keys are contiguous) ---
    Frag aP;
    aP.u[0] = *reinterpret_cast<const uint4*>(&Ps[wid][c16][g * 8]);
    aP.u[1] = *reinterpret_cast<const uint4*>(&Ps[wid][c16][16 + g * 8]);
    #pragma unroll
    for (int d = 0; d < 4; d++) {
      Frag bV;
      const _Float16* vp =
          vt + ((size_t)nh * HD + d * 16 + c16) * SEQ + kb + g * 8;
      bV.u[0] = *reinterpret_cast<const uint4*>(vp);
      bV.u[1] = *reinterpret_cast<const uint4*>(vp + 16);
      oacc[d] = __builtin_amdgcn_wmma_f32_16x16x32_f16(
          false, aP.v, false, bV.v, (short)0, oacc[d], false, false);
    }
    __syncthreads();
  }

  // Normalize and write o as f16 [n][s][768] at columns h*64 + ...
  #pragma unroll
  for (int d = 0; d < 4; d++)
    #pragma unroll
    for (int j = 0; j < 8; j++) {
      const int qi = qbase + rowadd + j;
      const float val = oacc[d][j] / lj[j];
      o[((size_t)n_ * SEQ + qi) * DIMM + h * HD + d * 16 + c16] = (_Float16)val;
    }
}

// ---------------------------------------------------------------------------
// Launcher.
// Inputs: 0=x 1=pos 2=padding_mask 3=ln_g 4=ln_b 5=w_qkv 6=b_qkv 7=w_out 8=b_out
// Workspace layout (f16 buffers, 256B-multiple sections):
//   xn[8192*768], wqkvT[2304*768], woutT[768*768],
//   q[96*1024*64], k[96*1024*64], vT[96*64*1024], o[8192*768]
// ---------------------------------------------------------------------------
extern "C" void kernel_launch(void* const* d_in, const int* in_sizes, int n_in,
                              void* d_out, int out_size, void* d_ws, size_t ws_size,
                              hipStream_t stream) {
  (void)in_sizes; (void)n_in; (void)out_size; (void)ws_size;
  const float* x     = (const float*)d_in[0];
  const float* ln_g  = (const float*)d_in[3];
  const float* ln_b  = (const float*)d_in[4];
  const float* w_qkv = (const float*)d_in[5];
  const float* b_qkv = (const float*)d_in[6];
  const float* w_out = (const float*)d_in[7];
  const float* b_out = (const float*)d_in[8];
  float* outp = (float*)d_out;

  char* ws = (char*)d_ws;
  const size_t szAct = (size_t)NB * SEQ * DIMM * 2;        // 12,582,912
  const size_t szWq  = (size_t)DIMM * QKVN * 2;            //  3,538,944
  const size_t szWo  = (size_t)DIMM * DIMM * 2;            //  1,179,648
  _Float16* xn    = (_Float16*)(ws);
  _Float16* wqkvT = (_Float16*)(ws + szAct);
  _Float16* woutT = (_Float16*)(ws + szAct + szWq);
  _Float16* qh    = (_Float16*)(ws + szAct + szWq + szWo);
  _Float16* kh    = (_Float16*)(ws + szAct + szWq + szWo + szAct);
  _Float16* vtp   = (_Float16*)(ws + szAct + szWq + szWo + 2 * szAct);
  _Float16* oh    = (_Float16*)(ws + szAct + szWq + szWo + 3 * szAct);

  ln_kernel<<<NB * SEQ, 256, 0, stream>>>(x, ln_g, ln_b, xn);
  cvtT_kernel<<<1024, 256, 0, stream>>>(w_qkv, wqkvT, DIMM, QKVN);
  cvtT_kernel<<<512, 256, 0, stream>>>(w_out, woutT, DIMM, DIMM);
  gemm_kernel<0><<<dim3(QKVN / 128, (NB * SEQ) / 128), 256, 0, stream>>>(
      xn, wqkvT, b_qkv, nullptr, nullptr, qh, kh, vtp);
  rope_kernel<<<(NB * NHEAD * SEQ * 16) / 256, 256, 0, stream>>>(qh, kh);
  attn_kernel<<<NB * NHEAD * (SEQ / 64), 128, 0, stream>>>(qh, kh, vtp, oh);
  gemm_kernel<1><<<dim3(DIMM / 128, (NB * SEQ) / 128), 256, 0, stream>>>(
      oh, woutT, b_out, x, outp, nullptr, nullptr, nullptr);
}
